// SeqRegressor_28415503631017
// MI455X (gfx1250) — matrix operand order repaired and checked by reading, not compiled, not run on hardware
//
#include <hip/hip_runtime.h>

// Only t = 0..6 influence the output (y[6] is all that is consumed downstream),
// so the 512-step LSTM collapses to a 7-step fused recurrence.
//
// Gate GEMMs run on v_wmma_f32_16x16x32_f16 (f16 in, f32 accumulate):
//   layer1: [x_t | h1] (K=43 -> pad 64)  x  [W_ih0 ; W_hh0]^T  (64 x 80)
//   layer2: [h1  | h2] (K=40 -> pad 64)  x  [W_ih1 ; W_hh1]^T  (64 x 80)
// Each wave owns a 16-row batch tile; 5 N-tiles x 2 K-chunks = 10 WMMA per
// layer-step. Biases are folded into the WMMA C accumulator.

typedef __attribute__((ext_vector_type(16))) _Float16 v16h;
typedef __attribute__((ext_vector_type(8)))  _Float16 v8h;
typedef __attribute__((ext_vector_type(8)))  float    v8f;

#define T_STEPS 7
#define B_TOT   4096
#define I_DIM   23
#define H_DIM   20
#define G_DIM   80          // 4*H
#define K_PAD   64
#define WAVES   4
#define ROWS_PER_BLOCK (WAVES * 16)   // 64
#define NBLOCKS (B_TOT / ROWS_PER_BLOCK)

__device__ __forceinline__ float sigmoidf_(float x) {
    return 1.0f / (1.0f + __expf(-x));
}

// A fragment for v_wmma_f32_16x16x32_f16, per ISA 7.12.2 (16-bit A 16x32):
//   lanes 0-15 : M=lane,   halves 0..7 -> K=0..7,  halves 8..15 -> K=16..23
//   lanes 16-31: M=lane-16,halves 0..7 -> K=8..15, halves 8..15 -> K=24..31
__device__ __forceinline__ v16h load_a_frag(const _Float16* rowPtr, int kc, int lane) {
    const int koff = kc * 32 + ((lane & 16) ? 8 : 0);
    v8h lo = *(const v8h*)(rowPtr + koff);
    v8h hi = *(const v8h*)(rowPtr + koff + 16);
    v16h r;
#pragma unroll
    for (int i = 0; i < 8; ++i) { r[i] = lo[i]; r[8 + i] = hi[i]; }
    return r;
}

__global__ __launch_bounds__(WAVES * 32)
void SeqRegressor_28415503631017_kernel(
    const float* __restrict__ x,      // [T,B,I]
    const float* __restrict__ h0,     // [2,B,H]
    const float* __restrict__ c0,     // [2,B,H]
    const float* __restrict__ W_ih0,  // [80,23]
    const float* __restrict__ W_hh0,  // [80,20]
    const float* __restrict__ b_ih0,  // [80]
    const float* __restrict__ b_hh0,  // [80]
    const float* __restrict__ W_ih1,  // [80,20]
    const float* __restrict__ W_hh1,  // [80,20]
    const float* __restrict__ b_ih1,  // [80]
    const float* __restrict__ b_hh1,  // [80]
    const float* __restrict__ W_lin,  // [1,20]
    const float* __restrict__ b_lin,  // [1]
    float* __restrict__ out)          // [B,1]
{
    // Weights pre-swizzled into B-fragment order: [ntile][kchunk][lane][16 halves],
    // so each lane's B fragment is one contiguous 32B LDS read.
    __shared__ __align__(32) _Float16 sW1[5][2][32][16];
    __shared__ __align__(32) _Float16 sW2[5][2][32][16];
    __shared__ float sB1[G_DIM];
    __shared__ float sB2[G_DIM];
    __shared__ __align__(16) _Float16 sA1[WAVES][16][K_PAD];  // [x_t | h1 | 0]
    __shared__ __align__(16) _Float16 sA2[WAVES][16][K_PAD];  // [h1  | h2 | 0]
    __shared__ float sG[WAVES][16][G_DIM];                    // gate staging

    const int tid  = threadIdx.x;
    const int wave = tid >> 5;
    const int lane = tid & 31;
    const int rowBase = blockIdx.x * ROWS_PER_BLOCK + wave * 16;

    // ---- stage weights (once per block), combined + padded + fp16, frag order ----
    for (int e = tid; e < 5 * 2 * 512; e += blockDim.x) {
        const int nt  = e / 1024;
        const int rem = e % 1024;
        const int kc  = rem / 512;
        const int l   = (rem % 512) / 16;
        const int hh  = e % 16;
        const int K = kc * 32 + ((l < 16) ? 0 : 8) + ((hh < 8) ? hh : hh + 8);   // B frag: lanes 0-15 K=0..7,16..23; 16-31 K=8..15,24..31
        const int N = nt * 16 + (l & 15);
        float w1, w2;
        if (K < I_DIM)              w1 = W_ih0[N * I_DIM + K];
        else if (K < I_DIM + H_DIM) w1 = W_hh0[N * H_DIM + (K - I_DIM)];
        else                        w1 = 0.0f;
        if (K < H_DIM)              w2 = W_ih1[N * H_DIM + K];
        else if (K < 2 * H_DIM)     w2 = W_hh1[N * H_DIM + (K - H_DIM)];
        else                        w2 = 0.0f;
        sW1[nt][kc][l][hh] = (_Float16)w1;
        sW2[nt][kc][l][hh] = (_Float16)w2;
    }
    for (int n = tid; n < G_DIM; n += blockDim.x) {
        sB1[n] = b_ih0[n] + b_hh0[n];
        sB2[n] = b_ih1[n] + b_hh1[n];
    }

    // ---- zero K-padding and seed initial hidden state into A staging ----
    for (int e = lane; e < 16 * K_PAD; e += 32) {
        const int r = e >> 6, k = e & 63;
        if (k >= I_DIM + H_DIM) sA1[wave][r][k] = (_Float16)0.0f;
        if (k >= 2 * H_DIM)     sA2[wave][r][k] = (_Float16)0.0f;
    }
    for (int e = lane; e < 16 * H_DIM; e += 32) {
        const int r = e / H_DIM, j = e % H_DIM;
        const int row = rowBase + r;
        sA1[wave][r][I_DIM + j] = (_Float16)h0[row * H_DIM + j];
        sA2[wave][r][H_DIM + j] = (_Float16)h0[B_TOT * H_DIM + row * H_DIM + j];
    }

    // each lane owns 10 of the 16x20 state elements of its wave tile
    float c1r[10], c2r[10], h2r[10];
#pragma unroll
    for (int q = 0; q < 10; ++q) {
        const int idx = lane * 10 + q;
        const int r = idx / H_DIM, j = idx % H_DIM;
        const int row = rowBase + r;
        c1r[q] = c0[row * H_DIM + j];
        c2r[q] = c0[B_TOT * H_DIM + row * H_DIM + j];
        h2r[q] = h0[B_TOT * H_DIM + row * H_DIM + j];
    }
    __syncthreads();

    for (int t = 0; t < T_STEPS; ++t) {
        // stage x[t] tile (fp32 -> fp16) into A1 cols 0..22
        for (int e = lane; e < 16 * I_DIM; e += 32) {
            const int r = e / I_DIM, ci = e % I_DIM;
            sA1[wave][r][ci] =
                (_Float16)x[((size_t)t * B_TOT + (rowBase + r)) * (size_t)I_DIM + ci];
        }
        if (t + 1 < T_STEPS) {  // global_prefetch_b8 for next timestep's tile
            __builtin_prefetch(x + ((size_t)(t + 1) * B_TOT + rowBase) * (size_t)I_DIM, 0, 1);
        }
        __syncthreads();

        // ---------------- layer 1 gate GEMM : 16x80 += A1(16x64) x W1(64x80) ----
        {
            const _Float16* rowPtr = &sA1[wave][lane & 15][0];
            const v16h a0 = load_a_frag(rowPtr, 0, lane);
            const v16h a1 = load_a_frag(rowPtr, 1, lane);
            const int m0 = (lane & 16) ? 8 : 0;
#pragma unroll
            for (int n = 0; n < 5; ++n) {
                const float bv = sB1[n * 16 + (lane & 15)];
                v8f acc;
#pragma unroll
                for (int r = 0; r < 8; ++r) acc[r] = bv;
                const v16h b0 = *(const v16h*)&sW1[n][0][lane][0];
                const v16h b1 = *(const v16h*)&sW1[n][1][lane][0];
                acc = __builtin_amdgcn_wmma_f32_16x16x32_f16(false, a0, false, b0,
                                                             (short)0, acc, false, false);
                acc = __builtin_amdgcn_wmma_f32_16x16x32_f16(false, a1, false, b1,
                                                             (short)0, acc, false, false);
                const int nc = n * 16 + (lane & 15);
#pragma unroll
                for (int r = 0; r < 8; ++r) sG[wave][m0 + r][nc] = acc[r];
            }
        }
        __syncthreads();

        // layer 1 cell update (gate order i, f, g, o)
#pragma unroll
        for (int q = 0; q < 10; ++q) {
            const int idx = lane * 10 + q;
            const int r = idx / H_DIM, j = idx % H_DIM;
            const float i_ = sigmoidf_(sG[wave][r][j]);
            const float f_ = sigmoidf_(sG[wave][r][H_DIM + j]);
            const float g_ = tanhf(sG[wave][r][2 * H_DIM + j]);
            const float o_ = sigmoidf_(sG[wave][r][3 * H_DIM + j]);
            c1r[q] = f_ * c1r[q] + i_ * g_;
            const float h_ = o_ * tanhf(c1r[q]);
            sA1[wave][r][I_DIM + j] = (_Float16)h_;  // recurrent input, next t
            sA2[wave][r][j]         = (_Float16)h_;  // feeds layer 2 now
        }
        __syncthreads();

        // ---------------- layer 2 gate GEMM ------------------------------------
        {
            const _Float16* rowPtr = &sA2[wave][lane & 15][0];
            const v16h a0 = load_a_frag(rowPtr, 0, lane);
            const v16h a1 = load_a_frag(rowPtr, 1, lane);
            const int m0 = (lane & 16) ? 8 : 0;
#pragma unroll
            for (int n = 0; n < 5; ++n) {
                const float bv = sB2[n * 16 + (lane & 15)];
                v8f acc;
#pragma unroll
                for (int r = 0; r < 8; ++r) acc[r] = bv;
                const v16h b0 = *(const v16h*)&sW2[n][0][lane][0];
                const v16h b1 = *(const v16h*)&sW2[n][1][lane][0];
                acc = __builtin_amdgcn_wmma_f32_16x16x32_f16(false, a0, false, b0,
                                                             (short)0, acc, false, false);
                acc = __builtin_amdgcn_wmma_f32_16x16x32_f16(false, a1, false, b1,
                                                             (short)0, acc, false, false);
                const int nc = n * 16 + (lane & 15);
#pragma unroll
                for (int r = 0; r < 8; ++r) sG[wave][m0 + r][nc] = acc[r];
            }
        }
        __syncthreads();

        // layer 2 cell update
#pragma unroll
        for (int q = 0; q < 10; ++q) {
            const int idx = lane * 10 + q;
            const int r = idx / H_DIM, j = idx % H_DIM;
            const float i_ = sigmoidf_(sG[wave][r][j]);
            const float f_ = sigmoidf_(sG[wave][r][H_DIM + j]);
            const float g_ = tanhf(sG[wave][r][2 * H_DIM + j]);
            const float o_ = sigmoidf_(sG[wave][r][3 * H_DIM + j]);
            c2r[q] = f_ * c2r[q] + i_ * g_;
            const float h_ = o_ * tanhf(c2r[q]);
            h2r[q] = h_;
            sA2[wave][r][H_DIM + j] = (_Float16)h_;  // recurrent input, next t
        }
        __syncthreads();
    }

    // ---- head: relu(h2[6]) . W_lin + b_lin, then relu -------------------------
#pragma unroll
    for (int q = 0; q < 10; ++q) {
        const int idx = lane * 10 + q;
        const int r = idx / H_DIM, j = idx % H_DIM;
        float h_ = h2r[q];
        h_ = h_ > 0.0f ? h_ : 0.0f;
        sG[wave][r][j] = h_ * W_lin[j];
    }
    __syncthreads();
    if (lane < 16) {
        float s = b_lin[0];
#pragma unroll
        for (int j = 0; j < H_DIM; ++j) s += sG[wave][lane][j];
        s = s > 0.0f ? s : 0.0f;
        out[rowBase + lane] = s;
    }
}

extern "C" void kernel_launch(void* const* d_in, const int* in_sizes, int n_in,
                              void* d_out, int out_size, void* d_ws, size_t ws_size,
                              hipStream_t stream) {
    (void)in_sizes; (void)n_in; (void)out_size; (void)d_ws; (void)ws_size;
    const float* x     = (const float*)d_in[0];
    const float* h0    = (const float*)d_in[1];
    const float* c0    = (const float*)d_in[2];
    const float* W_ih0 = (const float*)d_in[3];
    const float* W_hh0 = (const float*)d_in[4];
    const float* b_ih0 = (const float*)d_in[5];
    const float* b_hh0 = (const float*)d_in[6];
    const float* W_ih1 = (const float*)d_in[7];
    const float* W_hh1 = (const float*)d_in[8];
    const float* b_ih1 = (const float*)d_in[9];
    const float* b_hh1 = (const float*)d_in[10];
    const float* W_lin = (const float*)d_in[11];
    const float* b_lin = (const float*)d_in[12];
    float* out = (float*)d_out;

    SeqRegressor_28415503631017_kernel<<<NBLOCKS, WAVES * 32, 0, stream>>>(
        x, h0, c0, W_ih0, W_hh0, b_ih0, b_hh0,
        W_ih1, W_hh1, b_ih1, b_hh1, W_lin, b_lin, out);
}
